// SoftmaxAttention_76381698392283
// MI455X (gfx1250) — matrix operand order repaired and checked by reading, not compiled, not run on hardware
//
#include <hip/hip_runtime.h>
#include <hip/hip_bf16.h>
#include <stdint.h>
#include <stddef.h>

// ---------------------------------------------------------------------------
// Problem constants (from reference): B=32, SP=SH=256, H=512, D=2H=1024,
// E=H/2=256, FD=4H=2048. All GEMM dims divisible by tile sizes below.
// ---------------------------------------------------------------------------
static constexpr int kB  = 32;
static constexpr int kS  = 256;
static constexpr int kH  = 512;
static constexpr int kD  = 1024;
static constexpr int kE  = 256;
static constexpr int kFD = 2048;

typedef __bf16 bf16;
typedef __attribute__((ext_vector_type(16))) __bf16       v16bf;
typedef __attribute__((ext_vector_type(8)))  float        v8f;
typedef __attribute__((ext_vector_type(4)))  unsigned int v4u;
typedef __attribute__((ext_vector_type(4)))  int          v4i;

union Frag16 { v16bf v; v4u q[2]; };

// CDNA5 async global->LDS path (gfx1250): direct-to-LDS DMA tracked by
// ASYNCcnt. Guarded so missing builtins fall back to reg-staged copies.
// Builtin signature (from hipcc diagnostics): arg0 = v4i* in global AS,
// arg1 = v4i* in LDS AS, arg2/arg3 = imm offset / cpol.
#if defined(__has_builtin)
#if __has_builtin(__builtin_amdgcn_global_load_async_to_lds_b128) && \
    __has_builtin(__builtin_amdgcn_s_wait_asynccnt)
#define USE_ASYNC_LDS 1
#endif
#endif
#ifndef USE_ASYNC_LDS
#define USE_ASYNC_LDS 0
#endif

#if USE_ASYNC_LDS
typedef __attribute__((address_space(1))) v4i* gv4p;   // global int4*
typedef __attribute__((address_space(3))) v4i* lv4p;   // LDS int4*
#endif

// ---------------------------------------------------------------------------
// Tiled NT GEMM, bf16 inputs, f32 accumulate via v_wmma_f32_16x16x32_bf16.
//   C[m,n] = sum_k A[m,k] * Bt[n,k]      (A: MxK row-major, Bt: NxK row-major)
// Block tile 128x256x32, 256 threads = 8 wave32 waves in a 2x4 grid; each
// wave owns a 64x64 tile = 4x4 WMMA fragments (16 WMMA : 16 ds_load_b128 per
// K-step). LDS rows padded to 40 bf16 (80B = 20 banks) -> conflict-free b128
// fragment reads. Async path double-buffers LDS slabs via
// global_load_async_to_lds_b128 + s_wait_asynccnt.
// ---------------------------------------------------------------------------
struct EpiParams {
  int mode;            // 0: f32 = acc*alpha   1: bf16 = acc*rowmask
                       // 2: bf16 = acc*fuseScalar + bias[n]
                       // 3: bf16 = relu(acc + bias[n])   4: f32 += acc
  int fuse_k;
  float alpha;
  const float* bias;     // length N
  const int*   rowmask;  // (B,S) int mask, may be null
  const float* s1;       // (B,S) importance (self)
  const float* s2;       // (B,S) importance (inter)
};

struct GemmArgs {
  const bf16* A;  long long sA; int lda;
  const bf16* Bt; long long sB; int ldb;
  void* C;        long long sC; int ldc;
  int M, N, K;
  EpiParams e;
};

__global__ __launch_bounds__(256) void gemm_nt_bf16(GemmArgs g) {
  constexpr int LROW  = 40;            // padded LDS row (bf16 elems)
  constexpr int ASLAB = 128 * LROW;    // 5,120 bf16
  constexpr int BSLAB = 256 * LROW;    // 10,240 bf16
#if USE_ASYNC_LDS
  __shared__ bf16 lA[2 * ASLAB];       // double-buffered: 60 KB total LDS
  __shared__ bf16 lB[2 * BSLAB];
#else
  __shared__ bf16 lA[ASLAB];           // single buffer + register prefetch
  __shared__ bf16 lB[BSLAB];
#endif

  const int tid  = threadIdx.x;
  const int lane = tid & 31;
  const int wid  = tid >> 5;           // 8 waves
  const int wm   = wid & 1;            // 2 waves along M (64 rows each)
  const int wn   = wid >> 1;           // 4 waves along N (64 cols each)
  const int bM   = blockIdx.y * 128;
  const int bN   = blockIdx.x * 256;
  const int bz   = blockIdx.z;

  const bf16* Ab = g.A  + (long long)bz * g.sA + (long long)bM * g.lda;
  const bf16* Bb = g.Bt + (long long)bz * g.sB + (long long)bN * g.ldb;

  // global->LDS staging map: A tile 128x32 (2 threads/row, 2 b128 each),
  // B tile 256x32 (1 thread/row, 4 b128 each) -> 6 b128 per thread per slab.
  const int ra = tid >> 1;             // A row 0..127
  const int ca = (tid & 1) * 16;       // A elem offset 0/16
  const int rb = tid;                  // B row 0..255

  v8f acc[4][4];
#pragma unroll
  for (int i = 0; i < 4; ++i)
#pragma unroll
    for (int j = 0; j < 4; ++j)
#pragma unroll
      for (int r = 0; r < 8; ++r) acc[i][j][r] = 0.0f;

  const int nk    = g.K / 32;
  const int khalf = lane >> 4;         // wave32 low/high half-lanes
  const int rl    = lane & 15;

  // 16 WMMAs on one K-slab. A frag: lane<16 K{0..7,16..23} (+8 for hi half);
  // B frag: same K order as A (lane<16 K{0..7,16..23}, +8 for hi half).
  auto compute = [&](int sa, int sb) {
    Frag16 fa[4];
#pragma unroll
    for (int i = 0; i < 4; ++i) {
      const int row = wm * 64 + i * 16 + rl;
      fa[i].q[0] = *(const v4u*)&lA[sa + row * LROW + khalf * 8];
      fa[i].q[1] = *(const v4u*)&lA[sa + row * LROW + khalf * 8 + 16];
    }
#pragma unroll
    for (int j = 0; j < 4; ++j) {
      Frag16 fb;
      const int row = wn * 64 + j * 16 + rl;
      fb.q[0] = *(const v4u*)&lB[sb + row * LROW + khalf * 8];
      fb.q[1] = *(const v4u*)&lB[sb + row * LROW + khalf * 8 + 16];
#pragma unroll
      for (int i = 0; i < 4; ++i)
        acc[i][j] = __builtin_amdgcn_wmma_f32_16x16x32_bf16(
            false, fa[i].v, false, fb.v, (short)0, acc[i][j], false, false);
    }
  };

#if USE_ASYNC_LDS
  // Issue one K-slab of async direct-to-LDS loads (6 x b128 per thread).
  auto issue = [&](int kk, int sl) {
    const bf16* pa = Ab + (long long)kk * 32;
    const bf16* pb = Bb + (long long)kk * 32;
    __builtin_amdgcn_global_load_async_to_lds_b128(
        (gv4p)(pa + (long long)ra * g.lda + ca),
        (lv4p)&lA[sl * ASLAB + ra * LROW + ca], 0, 0);
    __builtin_amdgcn_global_load_async_to_lds_b128(
        (gv4p)(pa + (long long)ra * g.lda + ca + 8),
        (lv4p)&lA[sl * ASLAB + ra * LROW + ca + 8], 0, 0);
#pragma unroll
    for (int q = 0; q < 4; ++q)
      __builtin_amdgcn_global_load_async_to_lds_b128(
          (gv4p)(pb + (long long)rb * g.ldb + q * 8),
          (lv4p)&lB[sl * BSLAB + rb * LROW + q * 8], 0, 0);
  };

  issue(0, 0);
  for (int kk = 0; kk < nk; ++kk) {
    if (kk + 1 < nk) {
      issue(kk + 1, (kk + 1) & 1);              // prefetch next slab
      __builtin_amdgcn_s_wait_asynccnt(6);      // older slab's 6 retired
    } else {
      __builtin_amdgcn_s_wait_asynccnt(0);
    }
    __syncthreads();
    compute((kk & 1) * ASLAB, (kk & 1) * BSLAB);
    __syncthreads();
  }
#else
  // Fallback: register-staged prefetch, single LDS buffer.
  v4u aa0 = *(const v4u*)(Ab + (long long)ra * g.lda + ca);
  v4u aa1 = *(const v4u*)(Ab + (long long)ra * g.lda + ca + 8);
  v4u bb0 = *(const v4u*)(Bb + (long long)rb * g.ldb + 0);
  v4u bb1 = *(const v4u*)(Bb + (long long)rb * g.ldb + 8);
  v4u bb2 = *(const v4u*)(Bb + (long long)rb * g.ldb + 16);
  v4u bb3 = *(const v4u*)(Bb + (long long)rb * g.ldb + 24);
  for (int kk = 0; kk < nk; ++kk) {
    *(v4u*)&lA[ra * LROW + ca]     = aa0;
    *(v4u*)&lA[ra * LROW + ca + 8] = aa1;
    *(v4u*)&lB[rb * LROW + 0]      = bb0;
    *(v4u*)&lB[rb * LROW + 8]      = bb1;
    *(v4u*)&lB[rb * LROW + 16]     = bb2;
    *(v4u*)&lB[rb * LROW + 24]     = bb3;
    __syncthreads();
    if (kk + 1 < nk) {
      const bf16* pa = Ab + (long long)(kk + 1) * 32;
      const bf16* pb = Bb + (long long)(kk + 1) * 32;
      aa0 = *(const v4u*)(pa + (long long)ra * g.lda + ca);
      aa1 = *(const v4u*)(pa + (long long)ra * g.lda + ca + 8);
      bb0 = *(const v4u*)(pb + (long long)rb * g.ldb + 0);
      bb1 = *(const v4u*)(pb + (long long)rb * g.ldb + 8);
      bb2 = *(const v4u*)(pb + (long long)rb * g.ldb + 16);
      bb3 = *(const v4u*)(pb + (long long)rb * g.ldb + 24);
    }
    compute(0, 0);
    __syncthreads();
  }
#endif

  // Epilogue. C fragment layout: VGPR r -> M = r (+8 for lanes 16-31), N = rl.
  const long long cb = (long long)bz * g.sC;
  const int mhalf = khalf * 8;
#pragma unroll
  for (int i = 0; i < 4; ++i) {
    const int m0 = bM + wm * 64 + i * 16 + mhalf;
#pragma unroll
    for (int j = 0; j < 4; ++j) {
      const int n = bN + wn * 64 + j * 16 + rl;
#pragma unroll
      for (int r = 0; r < 8; ++r) {
        const int m = m0 + r;
        const float v = acc[i][j][r];
        const long long idx = cb + (long long)m * g.ldc + n;
        switch (g.e.mode) {
          case 0:
            ((float*)g.C)[idx] = v * g.e.alpha;
            break;
          case 1: {
            const float msk = g.e.rowmask ? (float)g.e.rowmask[bz * kS + m] : 1.0f;
            ((bf16*)g.C)[idx] = (bf16)(v * msk);
          } break;
          case 2: {
            const float s1 = g.e.s1[bz * kS + m];
            const float s2 = g.e.s2[bz * kS + m];
            float a;
            switch (g.e.fuse_k) {
              case 0: a = s1 * s2;              break;
              case 1: a = s1 + s2;              break;
              case 2: a = fmaxf(s1, s2);        break;
              case 4: a = s1 * s2 + 1.0f;       break;
              case 5: a = s1 + s2 + 1.0f;       break;
              case 6: a = fmaxf(s1, s2) + 1.0f; break;
              default: a = 1.0f;                break;   // k=3 (mx), k=7 (mx+x)
            }
            ((bf16*)g.C)[idx] = (bf16)(v * a + g.e.bias[n]);
          } break;
          case 3:
            ((bf16*)g.C)[idx] = (bf16)fmaxf(v + g.e.bias[n], 0.0f);
            break;
          default:  // 4: accumulate f32
            ((float*)g.C)[idx] += v;
            break;
        }
      }
    }
  }
}

// ---------------------------------------------------------------------------
// Elementwise / reduction kernels
// ---------------------------------------------------------------------------
__global__ void transpose_f32_to_bf16(const float* in, bf16* out, int R, int C) {
  const long long total = (long long)R * C;
  const long long base  = (long long)blockIdx.y * total;   // batch (for W8)
  const long long idx   = (long long)blockIdx.x * blockDim.x + threadIdx.x;
  if (idx >= total) return;
  const int r = (int)(idx / C);
  const int c = (int)(idx % C);
  out[base + (long long)c * R + r] = (bf16)in[base + idx];
}

__global__ void conv_x(const float* x, bf16* xb, bf16* xbt) {
  const long long idx = (long long)blockIdx.x * blockDim.x + threadIdx.x;
  const int d  = (int)(idx % kD);
  const long long bs = idx / kD;          // b*kS + s
  const int s  = (int)(bs % kS);
  const int b  = (int)(bs / kS);
  const float v = x[idx];
  xb[idx] = (bf16)v;
  xbt[((long long)b * kD + d) * kS + s] = (bf16)v;
}

// One 256-wide block per softmax row. Reads sim with (rs,cs) strides so the
// same kernel handles the transposed (hp) direction. Mask semantics match the
// reference exactly: t = sim*scale*m; q = softmax(t); r = q*m; r/(sum r + eps).
__global__ void softmax_rows(const float* sim, const int* mask, bf16* outp,
                             int rs, int cs, float scale, int masked) {
  __shared__ float red[kS];
  const int b = blockIdx.y, row = blockIdx.x, c = threadIdx.x;
  const float* sp = sim + (long long)b * kS * kS;
  const float mv = masked ? (float)mask[b * kS + c] : 1.0f;
  const float t  = sp[(long long)row * rs + (long long)c * cs] * scale * mv;
  red[c] = t; __syncthreads();
  for (int st = kS / 2; st > 0; st >>= 1) { if (c < st) red[c] = fmaxf(red[c], red[c + st]); __syncthreads(); }
  const float mx = red[0]; __syncthreads();
  const float e = __expf(t - mx);
  red[c] = e; __syncthreads();
  for (int st = kS / 2; st > 0; st >>= 1) { if (c < st) red[c] += red[c + st]; __syncthreads(); }
  const float Se = red[0]; __syncthreads();
  float o = e / Se;
  if (masked) {
    const float r = o * mv;
    red[c] = r; __syncthreads();
    for (int st = kS / 2; st > 0; st >>= 1) { if (c < st) red[c] += red[c + st]; __syncthreads(); }
    o = r / (red[0] + 1e-13f);
  }
  outp[((long long)b * kS + row) * kS + c] = (bf16)o;
}

// Column sums (attn.sum(axis=-2)) -> importance vectors. Coalesced reads.
__global__ void colsum(const bf16* attn, float* o) {
  const int b = blockIdx.x, c = threadIdx.x;
  const bf16* a = attn + (long long)b * kS * kS;
  float s = 0.0f;
  for (int r = 0; r < kS; ++r) s += (float)a[(long long)r * kS + c];
  o[b * kS + c] = s;
}

// mx = max(x*s1, x*s2) elementwise; addx=1 gives feature 7: mx + x.
__global__ void make_mx(const float* x, const float* s1, const float* s2,
                        bf16* out, int addx) {
  const long long idx = (long long)blockIdx.x * blockDim.x + threadIdx.x;
  const long long bs  = idx / kD;
  const float xv = x[idx];
  float v = fmaxf(xv * s1[bs], xv * s2[bs]);
  if (addx) v += xv;
  out[idx] = (bf16)v;
}

__global__ void prod_sub(const float* x, const bf16* att, bf16* xa, bf16* xs) {
  const long long idx = (long long)blockIdx.x * blockDim.x + threadIdx.x;
  const float a = (float)att[idx];
  const float v = x[idx];
  xa[idx] = (bf16)(v * a);
  xs[idx] = (bf16)(v - a);
}

__global__ void bias_relu_f32(float* o, const float* bias, long long total, int n) {
  const long long idx = (long long)blockIdx.x * blockDim.x + threadIdx.x;
  if (idx >= total) return;
  o[idx] = fmaxf(o[idx] + bias[idx % n], 0.0f);
}

// ---------------------------------------------------------------------------
// Host orchestration
// ---------------------------------------------------------------------------
extern "C" void kernel_launch(void* const* d_in, const int* in_sizes, int n_in,
                              void* d_out, int out_size, void* d_ws, size_t ws_size,
                              hipStream_t stream) {
  (void)in_sizes; (void)n_in; (void)out_size; (void)ws_size;
  const float* P  = (const float*)d_in[0];
  const float* Hy = (const float*)d_in[1];
  const int*   pm = (const int*)d_in[2];
  const int*   hm = (const int*)d_in[3];
  const float* W8 = (const float*)d_in[4];
  const float* B8 = (const float*)d_in[5];
  const float* Wl = (const float*)d_in[6];
  const float* bl = (const float*)d_in[7];
  const float* Wp = (const float*)d_in[8];
  const float* bp = (const float*)d_in[9];
  float* out = (float*)d_out;

  // Bump allocator over d_ws (total ~280 MB).
  char* wptr = (char*)d_ws;
  auto alloc = [&](long long elems, size_t esz) -> void* {
    void* p = (void*)wptr;
    size_t bytes = ((size_t)elems * esz + 255) & ~(size_t)255;
    wptr += bytes;
    return p;
  };
  const long long XE = (long long)kB * kS * kD;   // 8,388,608
  const long long SE = (long long)kB * kS * kS;   // 2,097,152
  bf16*  Pb    = (bf16*)alloc(XE, 2);
  bf16*  Hb    = (bf16*)alloc(XE, 2);
  bf16*  PbT   = (bf16*)alloc(XE, 2);             // (B, D, S)
  bf16*  HbT   = (bf16*)alloc(XE, 2);
  float* simPH = (float*)alloc(SE, 4);
  float* simPP = (float*)alloc(SE, 4);
  float* simHH = (float*)alloc(SE, 4);
  bf16*  phb   = (bf16*)alloc(SE, 2);
  bf16*  hpb   = (bf16*)alloc(SE, 2);
  bf16*  spb   = (bf16*)alloc(SE, 2);
  bf16*  shb   = (bf16*)alloc(SE, 2);
  float* impP  = (float*)alloc(kB * kS, 4);
  float* impH  = (float*)alloc(kB * kS, 4);
  float* iimpP = (float*)alloc(kB * kS, 4);
  float* iimpH = (float*)alloc(kB * kS, 4);
  bf16*  mxb   = (bf16*)alloc(XE, 2);
  bf16*  catb  = (bf16*)alloc((long long)kB * kS * kFD, 2);
  bf16*  attb  = (bf16*)alloc(XE, 2);
  bf16*  selfb = (bf16*)alloc(XE, 2);
  bf16*  xattb = (bf16*)alloc(XE, 2);
  bf16*  xsubb = (bf16*)alloc(XE, 2);
  bf16*  fuseb = (bf16*)alloc((long long)kB * kS * kFD, 2);
  bf16*  W8t   = (bf16*)alloc(8LL * kD * kE, 2);  // (8, E, D)
  bf16*  Wlt   = (bf16*)alloc((long long)kFD * kFD, 2);     // (FD, 8E)
  bf16*  Wpt   = (bf16*)alloc(7LL * kD * kH, 2);            // (H, 7D)

  const float scale = 0.03125f;  // 1/sqrt(D)
  const unsigned gx = (unsigned)(XE / 256);

  auto gemm = [&](const bf16* A, long long sA, int lda,
                  const bf16* Bt, long long sB, int ldb,
                  void* C, long long sC, int ldc,
                  int M, int N, int K, EpiParams e) {
    GemmArgs g{A, sA, lda, Bt, sB, ldb, C, sC, ldc, M, N, K, e};
    gemm_nt_bf16<<<dim3(N / 256, M / 128, kB), 256, 0, stream>>>(g);
  };

  // --- weights -> transposed bf16; inputs -> bf16 (+ transposed copies) ---
  transpose_f32_to_bf16<<<dim3((kD * kE) / 256, 8), 256, 0, stream>>>(W8, W8t, kD, kE);
  transpose_f32_to_bf16<<<dim3((kFD * kFD) / 256, 1), 256, 0, stream>>>(Wl, Wlt, 8 * kE, kFD);
  transpose_f32_to_bf16<<<dim3((7 * kD * kH) / 256, 1), 256, 0, stream>>>(Wp, Wpt, 7 * kD, kH);
  conv_x<<<dim3(gx), 256, 0, stream>>>(P,  Pb, PbT);
  conv_x<<<dim3(gx), 256, 0, stream>>>(Hy, Hb, HbT);

  // --- similarity GEMMs (NT: A row-major, Bt row-major, K contiguous) ---
  EpiParams e0{0, 0, 1.0f, nullptr, nullptr, nullptr, nullptr};
  gemm(Pb, (long long)kS * kD, kD, Hb, (long long)kS * kD, kD,
       simPH, (long long)kS * kS, kS, kS, kS, kD, e0);
  gemm(Pb, (long long)kS * kD, kD, Pb, (long long)kS * kD, kD,
       simPP, (long long)kS * kS, kS, kS, kS, kD, e0);
  gemm(Hb, (long long)kS * kD, kD, Hb, (long long)kS * kD, kD,
       simHH, (long long)kS * kS, kS, kS, kS, kD, e0);

  // --- softmaxes (ph, hp masked; sp, sh plain) + importance column sums ---
  softmax_rows<<<dim3(kS, kB), kS, 0, stream>>>(simPH, hm, phb, kS, 1, scale, 1);
  softmax_rows<<<dim3(kS, kB), kS, 0, stream>>>(simPH, pm, hpb, 1, kS, scale, 1);
  softmax_rows<<<dim3(kS, kB), kS, 0, stream>>>(simPP, nullptr, spb, kS, 1, scale, 0);
  softmax_rows<<<dim3(kS, kB), kS, 0, stream>>>(simHH, nullptr, shb, kS, 1, scale, 0);
  colsum<<<dim3(kB), kS, 0, stream>>>(phb, iimpH);   // inter_imp_h = ph.sum(-2)
  colsum<<<dim3(kB), kS, 0, stream>>>(hpb, iimpP);   // inter_imp_p = hp.sum(-2)
  colsum<<<dim3(kB), kS, 0, stream>>>(spb, impP);
  colsum<<<dim3(kB), kS, 0, stream>>>(shb, impH);

  // --- per-side pipeline (buffers reused between sides) ---
  for (int side = 0; side < 2; ++side) {
    const float* X      = side ? Hy  : P;
    const bf16*  Xb     = side ? Hb  : Pb;
    const bf16*  SelfT  = side ? HbT : PbT;   // self_x = self_attn @ x
    const bf16*  PartT  = side ? PbT : HbT;   // att_h = hp @ P, att_p = ph @ H
    const bf16*  aInter = side ? hpb : phb;
    const bf16*  aSelf  = side ? shb : spb;
    const int*   msk    = side ? hm  : pm;
    const float* s1     = side ? impH  : impP;
    const float* s2     = side ? iimpH : iimpP;
    float* outS = out + (long long)side * kB * kS * kH;

    // att (row-masked) and self projections
    EpiParams em{1, 0, 1.0f, nullptr, msk, nullptr, nullptr};
    gemm(aInter, (long long)kS * kS, kS, PartT, (long long)kD * kS, kS,
         attb, (long long)kS * kD, kD, kS, kD, kS, em);
    EpiParams es{1, 0, 1.0f, nullptr, nullptr, nullptr, nullptr};
    gemm(aSelf, (long long)kS * kS, kS, SelfT, (long long)kD * kS, kS,
         selfb, (long long)kS * kD, kD, kS, kD, kS, es);
    prod_sub<<<dim3(gx), 256, 0, stream>>>(X, attb, xattb, xsubb);

    // fuse: 6 features are per-row scalar multiples of x -> scalar in epilogue
    for (int k = 0; k < 8; ++k) {
      if (k == 3 || k == 7) continue;
      EpiParams ef{2, k, 1.0f, B8 + k * kE, nullptr, s1, s2};
      gemm(Xb, (long long)kS * kD, kD, W8t + (long long)k * kE * kD, 0, kD,
           catb + k * kE, (long long)kS * kFD, kFD, kS, kE, kD, ef);
    }
    make_mx<<<dim3(gx), 256, 0, stream>>>(X, s1, s2, mxb, 0);
    { EpiParams ef{2, 3, 1.0f, B8 + 3 * kE, nullptr, s1, s2};
      gemm(mxb, (long long)kS * kD, kD, W8t + 3LL * kE * kD, 0, kD,
           catb + 3 * kE, (long long)kS * kFD, kFD, kS, kE, kD, ef); }
    make_mx<<<dim3(gx), 256, 0, stream>>>(X, s1, s2, mxb, 1);
    { EpiParams ef{2, 7, 1.0f, B8 + 7 * kE, nullptr, s1, s2};
      gemm(mxb, (long long)kS * kD, kD, W8t + 7LL * kE * kD, 0, kD,
           catb + 7 * kE, (long long)kS * kFD, kFD, kS, kE, kD, ef); }

    // x_all = relu(cat @ Wl + bl)
    EpiParams er{3, 0, 1.0f, bl, nullptr, nullptr, nullptr};
    gemm(catb, (long long)kS * kFD, kFD, Wlt, 0, kFD,
         fuseb, (long long)kS * kFD, kFD, kS, kFD, kFD, er);

    // out = enh @ Wp decomposed over the 6 K-blocks of Wp (first writes, rest +=)
    struct Comp { const bf16* A; long long sA; int lda; int off; int K; };
    const Comp comps[6] = {
      {Xb,    (long long)kS * kD,  kD,  0,            kD },
      {attb,  (long long)kS * kD,  kD,  kD,           kD },
      {selfb, (long long)kS * kD,  kD,  2 * kD,       kD },
      {fuseb, (long long)kS * kFD, kFD, 3 * kD,       kFD},
      {xattb, (long long)kS * kD,  kD,  3 * kD + kFD, kD },
      {xsubb, (long long)kS * kD,  kD,  4 * kD + kFD, kD },
    };
    for (int c = 0; c < 6; ++c) {
      EpiParams ec{c == 0 ? 0 : 4, 0, 1.0f, nullptr, nullptr, nullptr, nullptr};
      gemm(comps[c].A, comps[c].sA, comps[c].lda,
           Wpt + comps[c].off, 0, 7 * kD,
           outS, (long long)kS * kH, kH, kS, kH, comps[c].K, ec);
    }
  }

  // final bias + relu over both halves of d_out
  const long long total = 2LL * kB * kS * kH;
  bias_relu_f32<<<dim3((unsigned)(total / 256)), 256, 0, stream>>>(out, bp, total, kH);
}
